// DCT_base_Rec_Module_36421322670615
// MI455X (gfx1250) — matrix operand order repaired and checked by reading, not compiled, not run on hardware
//
#include <hip/hip_runtime.h>
#include <hip/hip_bf16.h>
#include <math.h>

// ---------------------------------------------------------------------------
// DCT_base_Rec_Module for MI455X (gfx1250, wave32, WMMA).
//   pass 1: per-patch grade via v_wmma_f32_16x16x32_f16  (f16 in, f32 acc)
//           - D fragments via hardware v_cos_f32 (__cosf): cast to f16 anyway
//           - grade via hardware v_log_f32 (__logf): ranking-only, f16 noise
//             already dominates
//   pass 2: top-2 / bottom-2 selection
//   pass 3: exact f32 reconstruction (f64-accurate D) of the 4 selected patches
// ---------------------------------------------------------------------------

typedef __attribute__((ext_vector_type(16))) _Float16 v16h;
typedef __attribute__((ext_vector_type(8)))  float    v8f;

#define WS_     32
#define STRIDE_ 16
#define HW_     2048
#define OH_     127
#define NPATCH_ (OH_ * OH_)   // 16129
#define CH_     3
#define PI_     3.14159265358979323846

// D[i][j], fast path: values feed f16 WMMA fragments (f16 eps ~5e-4 >> __cosf err)
__device__ __forceinline__ float dct_fast(int i, int j) {
    if (i == 0) return 0.17677669529663687f;                 // sqrt(1/32)
    return 0.25f * __cosf(((float)j + 0.5f) * (float)PI_ * (float)i / 32.0f);
}
// D[i][j], f64-accurate (matches numpy construction; pass 3 output path only)
__device__ __forceinline__ float dct_f64(int i, int j) {
    if (i == 0) return 0.17677669529663687f;
    return (float)(0.25 * cos(((double)j + 0.5) * PI_ * (double)i / 32.0));
}

// ---------------------------------------------------------------------------
// Pass 1: one wave32 per patch; 24 WMMA / patch.
//
// Fragment layout assumptions (CDNA5 ISA §7.12.2, wave32):
//   A 16x32 f16 : lane m = ln&15, half h = ln>>4; v16h elem e -> K =
//                 (e<8 ? 8h+e : 16+8h+(e-8))
//   B 32x16 f16 : lane N = ln&15, half h = ln>>4; v16h elem e -> K as for A
//   C/D 16x16 f32: vgpr r, lane ln -> (M = r + 8h, N = ln&15)
// ---------------------------------------------------------------------------
__global__ __launch_bounds__(256) void grade_kernel(
    const float* __restrict__ x, const float* __restrict__ Wt,
    float* __restrict__ grade)
{
    __shared__ float Wl[CH_ * WS_ * WS_];                 // 12 KB
    __shared__ __align__(16) _Float16 ldsT[8][WS_ * WS_]; // 16 KB, per-wave slice

    for (int t = threadIdx.x; t < CH_ * WS_ * WS_; t += 256) Wl[t] = Wt[t];
    __syncthreads();

    const int wave = threadIdx.x >> 5;
    const int ln   = threadIdx.x & 31;
    const int l0   = blockIdx.x * 8 + wave;
    const int l    = (l0 < NPATCH_) ? l0 : (NPATCH_ - 1);  // clamp: EXEC stays all-1s
    const int ph   = l / OH_, pw = l % OH_;
    const int hsel = ln >> 4;
    const int mloc = ln & 15;

    // Constant DCT fragments: A = D (two 16-row blocks), B = D^T (two 16-col blocks)
    v16h aD[2], bDT[2];
#pragma unroll
    for (int e = 0; e < 16; ++e) {
        const int k = (e < 8) ? (8 * hsel + e) : (16 + 8 * hsel + (e - 8));
        aD[0][e]  = (_Float16)dct_fast(mloc,      k);
        aD[1][e]  = (_Float16)dct_fast(16 + mloc, k);
        bDT[0][e] = (_Float16)dct_fast(mloc,      k);   // D^T[k][mloc]      = D[mloc][k]
        bDT[1][e] = (_Float16)dct_fast(16 + mloc, k);   // D^T[k][16+mloc]
    }

    _Float16* tl = ldsT[wave];
    float g = 0.0f;

    for (int c = 0; c < CH_; ++c) {
        // lane owns patch column N = mloc (+16*bj); elem e -> K row in table order
        const float* col = x + (size_t)c * HW_ * HW_
                             + (size_t)(ph * STRIDE_) * HW_ + pw * STRIDE_ + mloc;

        v16h bP[2];
#pragma unroll
        for (int e = 0; e < 16; ++e) {
            const int k = (e < 8) ? (8 * hsel + e) : (16 + 8 * hsel + (e - 8));
            bP[0][e] = (_Float16)col[(size_t)k * HW_];
            bP[1][e] = (_Float16)col[(size_t)k * HW_ + 16];
        }

        // T = D @ p  (four 16x16 blocks, K = 32 in one WMMA each)
        v8f T[2][2];
#pragma unroll
        for (int bi = 0; bi < 2; ++bi)
#pragma unroll
            for (int bj = 0; bj < 2; ++bj) {
                v8f z = {};
                T[bi][bj] = __builtin_amdgcn_wmma_f32_16x16x32_f16(
                    false, aD[bi], false, bP[bj], (short)0, z, false, false);
            }

        // stage T (f16, row-major) in this wave's private LDS slice
#pragma unroll
        for (int bi = 0; bi < 2; ++bi)
#pragma unroll
            for (int bj = 0; bj < 2; ++bj)
#pragma unroll
                for (int r = 0; r < 8; ++r) {
                    const int rr = 16 * bi + r + 8 * hsel;
                    const int cc = 16 * bj + mloc;
                    tl[rr * 32 + cc] = (_Float16)T[bi][bj][r];
                }

        // re-read T in A-fragment layout (two 16B-aligned b128 per block)
        union AU { uint4 q[2]; v16h h; };
        v16h aT[2];
#pragma unroll
        for (int bi = 0; bi < 2; ++bi) {
            const int rr = 16 * bi + mloc;
            AU u;
            u.q[0] = *(const uint4*)(tl + rr * 32 + 8 * hsel);
            u.q[1] = *(const uint4*)(tl + rr * 32 + 16 + 8 * hsel);
            aT[bi] = u.h;
        }

        // S = T @ D^T  == x_dct block;  grade += log(|S|+1) * W
#pragma unroll
        for (int bi = 0; bi < 2; ++bi)
#pragma unroll
            for (int bj = 0; bj < 2; ++bj) {
                v8f z = {};
                v8f S = __builtin_amdgcn_wmma_f32_16x16x32_f16(
                    false, aT[bi], false, bDT[bj], (short)0, z, false, false);
#pragma unroll
                for (int r = 0; r < 8; ++r) {
                    const int rr = 16 * bi + r + 8 * hsel;
                    const int cc = 16 * bj + mloc;
                    g += __logf(__builtin_fabsf(S[r]) + 1.0f)
                         * Wl[(c * 32 + rr) * 32 + cc];
                }
            }
    }

    // wave32 reduction
#pragma unroll
    for (int off = 16; off > 0; off >>= 1) g += __shfl_xor(g, off, 32);
    if (ln == 0 && l0 < NPATCH_) grade[l0] = g;
}

// ---------------------------------------------------------------------------
// Pass 2: indices of 2 smallest / 2 largest grades (stable-argsort tiebreak)
// ---------------------------------------------------------------------------
__global__ __launch_bounds__(256) void select_kernel(
    const float* __restrict__ grade, int* __restrict__ sel)
{
    __shared__ float smn0[256], smn1[256], smx0[256], smx1[256];
    __shared__ int   imn0[256], imn1[256], imx0[256], imx1[256];

    float mn0 = INFINITY, mn1 = INFINITY, mx0 = -INFINITY, mx1 = -INFINITY;
    int i0 = 0x7fffffff, i1 = 0x7fffffff, j0 = -1, j1 = -1;

    for (int i = threadIdx.x; i < NPATCH_; i += 256) {
        const float v = grade[i];
        if (v < mn0 || (v == mn0 && i < i0)) { mn1 = mn0; i1 = i0; mn0 = v; i0 = i; }
        else if (v < mn1 || (v == mn1 && i < i1)) { mn1 = v; i1 = i; }
        if (v > mx0 || (v == mx0 && i > j0)) { mx1 = mx0; j1 = j0; mx0 = v; j0 = i; }
        else if (v > mx1 || (v == mx1 && i > j1)) { mx1 = v; j1 = i; }
    }
    smn0[threadIdx.x] = mn0; imn0[threadIdx.x] = i0;
    smn1[threadIdx.x] = mn1; imn1[threadIdx.x] = i1;
    smx0[threadIdx.x] = mx0; imx0[threadIdx.x] = j0;
    smx1[threadIdx.x] = mx1; imx1[threadIdx.x] = j1;
    __syncthreads();

    if (threadIdx.x == 0) {
        float gmn0 = INFINITY, gmn1 = INFINITY, gmx0 = -INFINITY, gmx1 = -INFINITY;
        int gi0 = 0x7fffffff, gi1 = 0x7fffffff, gj0 = -1, gj1 = -1;
        for (int t = 0; t < 256; ++t) {
            for (int s = 0; s < 2; ++s) {
                const float v = s ? smn1[t] : smn0[t];
                const int   i = s ? imn1[t] : imn0[t];
                if (v < gmn0 || (v == gmn0 && i < gi0)) { gmn1 = gmn0; gi1 = gi0; gmn0 = v; gi0 = i; }
                else if (v < gmn1 || (v == gmn1 && i < gi1)) { gmn1 = v; gi1 = i; }
            }
            for (int s = 0; s < 2; ++s) {
                const float v = s ? smx1[t] : smx0[t];
                const int   i = s ? imx1[t] : imx0[t];
                if (v > gmx0 || (v == gmx0 && i > gj0)) { gmx1 = gmx0; gj1 = gj0; gmx0 = v; gj0 = i; }
                else if (v > gmx1 || (v == gmx1 && i > gj1)) { gmx1 = v; gj1 = i; }
            }
        }
        sel[0] = gi0;   // x_minmin
        sel[1] = gj0;   // x_maxmax
        sel[2] = gi1;   // x_minmin1
        sel[3] = gj1;   // x_maxmax1
    }
}

// ---------------------------------------------------------------------------
// Pass 3: exact f32  y = D^T (D p D^T) D  for the 4 selected patches.
// One block per (selection, channel): 12 blocks total (tiny).
// ---------------------------------------------------------------------------
__global__ __launch_bounds__(256) void recon_kernel(
    const float* __restrict__ x, const int* __restrict__ sel,
    float* __restrict__ out)
{
    __shared__ float Ds[32][32], M0[32][32], M1[32][32];

    const int s = blockIdx.x / CH_;
    const int c = blockIdx.x % CH_;
    const int l = sel[s];
    const int ph = l / OH_, pw = l % OH_;

    for (int t = threadIdx.x; t < 1024; t += 256) {
        const int i = t >> 5, j = t & 31;
        Ds[i][j] = dct_f64(i, j);
        M0[i][j] = x[(size_t)c * HW_ * HW_
                     + (size_t)(ph * STRIDE_ + i) * HW_ + (pw * STRIDE_ + j)];
    }
    __syncthreads();

    // M1 = D @ M0
    for (int t = threadIdx.x; t < 1024; t += 256) {
        const int i = t >> 5, j = t & 31;
        float a = 0.0f;
#pragma unroll
        for (int k = 0; k < 32; ++k) a += Ds[i][k] * M0[k][j];
        M1[i][j] = a;
    }
    __syncthreads();
    // M0 = M1 @ D^T
    for (int t = threadIdx.x; t < 1024; t += 256) {
        const int i = t >> 5, j = t & 31;
        float a = 0.0f;
#pragma unroll
        for (int k = 0; k < 32; ++k) a += M1[i][k] * Ds[j][k];
        M0[i][j] = a;
    }
    __syncthreads();
    // M1 = D^T @ M0
    for (int t = threadIdx.x; t < 1024; t += 256) {
        const int i = t >> 5, j = t & 31;
        float a = 0.0f;
#pragma unroll
        for (int k = 0; k < 32; ++k) a += Ds[k][i] * M0[k][j];
        M1[i][j] = a;
    }
    __syncthreads();
    // out = M1 @ D
    for (int t = threadIdx.x; t < 1024; t += 256) {
        const int i = t >> 5, j = t & 31;
        float a = 0.0f;
#pragma unroll
        for (int k = 0; k < 32; ++k) a += M1[i][k] * Ds[k][j];
        out[((size_t)s * CH_ + c) * 1024 + t] = a;
    }
}

// ---------------------------------------------------------------------------
extern "C" void kernel_launch(void* const* d_in, const int* in_sizes, int n_in,
                              void* d_out, int out_size, void* d_ws, size_t ws_size,
                              hipStream_t stream) {
    const float* x  = (const float*)d_in[0];   // [3,2048,2048]
    const float* Wt = (const float*)d_in[1];   // [1,3,32,32]
    float* grade = (float*)d_ws;
    int*   sel   = (int*)((char*)d_ws + ((NPATCH_ * sizeof(float) + 255) & ~(size_t)255));

    grade_kernel<<<(NPATCH_ + 7) / 8, 256, 0, stream>>>(x, Wt, grade);
    select_kernel<<<1, 256, 0, stream>>>(grade, sel);
    recon_kernel<<<4 * CH_, 256, 0, stream>>>(x, sel, (float*)d_out);
}
